// BiLSTM_21878563405976
// MI455X (gfx1250) — matrix operand order repaired and checked by reading, not compiled, not run on hardware
//
#include <hip/hip_runtime.h>
#include <math.h>

typedef __bf16 bf16;
typedef __attribute__((ext_vector_type(16))) __bf16 v16bf;
typedef __attribute__((ext_vector_type(8)))  __bf16 v8bf;
typedef __attribute__((ext_vector_type(8)))  float   v8f;

#define A_DIM 512
#define T_DIM 128
#define I_DIM 768
#define H_DIM 256
#define G_DIM 1024   // 4*H
#define H2    512    // 2*H

// ---------- helpers ----------
__device__ __forceinline__ bf16 f2bf(float f) {
  unsigned u = __float_as_uint(f);
  u += 0x7FFFu + ((u >> 16) & 1u);                 // round-to-nearest-even
  unsigned short s = (unsigned short)(u >> 16);
  bf16 r;
  __builtin_memcpy(&r, &s, sizeof(r));
  return r;
}

__device__ __forceinline__ float sigm(float x) {
  return 1.0f / (1.0f + __expf(-x));
}

// ---------- conversion / prep kernels ----------
__global__ void k_cvt_bf16(const float* __restrict__ src, bf16* __restrict__ dst, int n) {
  int i = blockIdx.x * blockDim.x + threadIdx.x;
  int stride = gridDim.x * blockDim.x;
  for (; i < n; i += stride) dst[i] = f2bf(src[i]);
}

// strided-row f32 -> packed bf16 (for x chunks: src row stride = T*I)
__global__ void k_cvt_rows(const float* __restrict__ src, int ld,
                           bf16* __restrict__ dst, int rows, int cols) {
  int i = blockIdx.x * blockDim.x + threadIdx.x;
  int n = rows * cols;
  int stride = gridDim.x * blockDim.x;
  for (; i < n; i += stride) {
    int r = i / cols, c = i - r * cols;
    dst[(size_t)r * cols + c] = f2bf(src[(size_t)r * ld + c]);
  }
}

// w [rows, cols] -> wt [cols, rows]
__global__ void k_transpose(const float* __restrict__ w, float* __restrict__ wt,
                            int rows, int cols) {
  int i = blockIdx.x * blockDim.x + threadIdx.x;
  int n = rows * cols;
  int stride = gridDim.x * blockDim.x;
  for (; i < n; i += stride) {
    int r = i / cols, c = i - r * cols;
    wt[c * rows + r] = w[i];
  }
}

__global__ void k_zero(float* __restrict__ p, int n) {
  int i = blockIdx.x * blockDim.x + threadIdx.x;
  int stride = gridDim.x * blockDim.x;
  for (; i < n; i += stride) p[i] = 0.0f;
}

// ---------- WMMA GEMM:  C = A(bf16,[M,K],lda) * W(bf16,[N,K],ldw)^T + bias ----------
// Wave tile: 16M x 64N (1 A-fragment reused across 4 B-fragments / 4 accumulators).
// Block 64 threads = 2 waves -> block tile 16M x 128N.
// grid: (N/128, M/16, nz); blockIdx.z selects problem 0 / 1.
__global__ __launch_bounds__(64)
void k_gemm_wmma(const bf16* __restrict__ A0, const bf16* __restrict__ A1, int lda,
                 const bf16* __restrict__ W0, const bf16* __restrict__ W1, int ldw,
                 const float* __restrict__ bias0, const float* __restrict__ bias1,
                 float* __restrict__ C0, float* __restrict__ C1, int ldc, int K) {
  const int wave = threadIdx.x >> 5;
  const int lane = threadIdx.x & 31;
  const int half = lane >> 4;          // lane group 0 / 1
  const int l15  = lane & 15;
  const int n0 = (blockIdx.x * 2 + wave) * 64;
  const int m0 = blockIdx.y * 16;

  const bf16*  A    = blockIdx.z ? A1 : A0;
  const bf16*  W    = blockIdx.z ? W1 : W0;
  const float* bias = blockIdx.z ? bias1 : bias0;
  float*       C    = blockIdx.z ? C1 : C0;

  // A: lanes 0-15 hold K kbA+0..15 (contiguous, kbA = 16*half); row m0+l15
  const bf16* arow  = A + (size_t)(m0 + l15) * lda;
  const int   kbA   = half * 16;
  // B: lane = output column N, 16 contiguous K per half
  const bf16* wbase = W + (size_t)(n0 + l15) * ldw + half * 16;
  const size_t wstep = (size_t)16 * ldw;

  v8f acc0 = {}, acc1 = {}, acc2 = {}, acc3 = {};
  for (int k0 = 0; k0 < K; k0 += 32) {
    v16bf a;
    v8bf alo = *(const v8bf*)(arow + k0 + kbA);
    v8bf ahi = *(const v8bf*)(arow + k0 + 8 + kbA);
#pragma unroll
    for (int i = 0; i < 8; ++i) { a[i] = alo[i]; a[8 + i] = ahi[i]; }

    v16bf b0 = *(const v16bf*)(wbase + k0);
    v16bf b1 = *(const v16bf*)(wbase + wstep + k0);
    v16bf b2 = *(const v16bf*)(wbase + 2 * wstep + k0);
    v16bf b3 = *(const v16bf*)(wbase + 3 * wstep + k0);

    acc0 = __builtin_amdgcn_wmma_f32_16x16x32_bf16(false, a, false, b0, (short)0, acc0, false, false);
    acc1 = __builtin_amdgcn_wmma_f32_16x16x32_bf16(false, a, false, b1, (short)0, acc1, false, false);
    acc2 = __builtin_amdgcn_wmma_f32_16x16x32_bf16(false, a, false, b2, (short)0, acc2, false, false);
    acc3 = __builtin_amdgcn_wmma_f32_16x16x32_bf16(false, a, false, b3, (short)0, acc3, false, false);
  }

  // C layout: VGPR r -> M = m0 + half*8 + r ; N = tile_base + l15
#pragma unroll
  for (int tile = 0; tile < 4; ++tile) {
    const v8f acc = tile == 0 ? acc0 : tile == 1 ? acc1 : tile == 2 ? acc2 : acc3;
    const int col = n0 + tile * 16 + l15;
    const float bv = bias[col];
#pragma unroll
    for (int r = 0; r < 8; ++r) {
      const int row = m0 + half * 8 + r;
      C[(size_t)row * ldc + col] = acc[r] + bv;
    }
  }
}

// ---------- sequential LSTM over one chunk (512 steps), one block per direction ----------
// xg_*   : [512, 1024] f32 precomputed  x@Wih^T + b
// whhT_* : [256, 1024] f32 (transposed so per-k loads are coalesced, L2-resident)
// hc     : per-chain state, chain c at hc + c*512: [0,256)=h, [256,512)=c
// y_bf   : [512, 512] bf16 chunk output, dir f -> cols [0,256), dir b -> cols [256,512)
__global__ __launch_bounds__(1024)
void k_lstm_chunk(const float* __restrict__ xg_f, const float* __restrict__ xg_b,
                  const float* __restrict__ whhT_f, const float* __restrict__ whhT_b,
                  float* __restrict__ hc, int chain_base,
                  bf16* __restrict__ y_bf) {
  const int dir = blockIdx.x;            // 0 = forward, 1 = backward (reversed in-chunk)
  const int j   = threadIdx.x;           // gate row 0..1023 (i | f | g | o blocks of 256)
  const float* xg   = dir ? xg_b   : xg_f;
  const float* whhT = dir ? whhT_b : whhT_f;
  float* st = hc + (size_t)(chain_base + dir) * 512;

  __shared__ float h_sh[H_DIM];
  __shared__ float c_sh[H_DIM];
  __shared__ float g_sh[G_DIM];

  if (j < H_DIM) { h_sh[j] = st[j]; c_sh[j] = st[H_DIM + j]; }
  __syncthreads();

  const float* wcol = whhT + j;          // column j, stride G_DIM over k
  for (int s = 0; s < A_DIM; ++s) {
    const int a = dir ? (A_DIM - 1 - s) : s;
    float acc = xg[(size_t)a * G_DIM + j];
#pragma unroll 8
    for (int k = 0; k < H_DIM; ++k)
      acc = fmaf(wcol[(size_t)k * G_DIM], h_sh[k], acc);
    g_sh[j] = acc;
    __syncthreads();                     // all reads of h_sh done, all gates visible
    if (j < H_DIM) {
      const float ig = sigm(g_sh[j]);
      const float fg = sigm(g_sh[H_DIM + j]);
      const float gg = tanhf(g_sh[2 * H_DIM + j]);
      const float og = sigm(g_sh[3 * H_DIM + j]);
      const float c  = fg * c_sh[j] + ig * gg;
      const float hh = og * tanhf(c);
      c_sh[j] = c;
      h_sh[j] = hh;
      y_bf[(size_t)a * H2 + dir * H_DIM + j] = f2bf(hh);
    }
    __syncthreads();                     // new h visible before next step's dot
  }
  if (j < H_DIM) { st[j] = h_sh[j]; st[H_DIM + j] = c_sh[j]; }
}

// ---------- classifier tail: logits = hdn @ w2^T + b2 ; softmax over 13 ----------
__global__ __launch_bounds__(32)
void k_head(const float* __restrict__ hdn, const float* __restrict__ w2,
            const float* __restrict__ b2, float* __restrict__ out) {
  const int a = blockIdx.x;
  const int t = threadIdx.x;
  __shared__ float lg[13];
  if (t < 13) {
    float acc = b2[t];
    const float* hr = hdn + (size_t)a * 128;
    const float* wr = w2 + (size_t)t * 128;
#pragma unroll 8
    for (int k = 0; k < 128; ++k) acc = fmaf(hr[k], wr[k], acc);
    lg[t] = acc;
  }
  __syncthreads();
  if (t == 0) {
    float mx = lg[0];
    for (int i = 1; i < 13; ++i) mx = fmaxf(mx, lg[i]);
    float sum = 0.0f;
    float e[13];
    for (int i = 0; i < 13; ++i) { e[i] = __expf(lg[i] - mx); sum += e[i]; }
    const float inv = 1.0f / sum;
    for (int i = 0; i < 13; ++i) out[(size_t)a * 13 + i] = e[i] * inv;
  }
}

// ---------- launch ----------
extern "C" void kernel_launch(void* const* d_in, const int* in_sizes, int n_in,
                              void* d_out, int out_size, void* d_ws, size_t ws_size,
                              hipStream_t stream) {
  (void)in_sizes; (void)n_in; (void)out_size; (void)ws_size;
  const float* x     = (const float*)d_in[0];
  const float* wih0f = (const float*)d_in[1];
  const float* whh0f = (const float*)d_in[2];
  const float* b0f   = (const float*)d_in[3];
  const float* wih0b = (const float*)d_in[4];
  const float* whh0b = (const float*)d_in[5];
  const float* b0b   = (const float*)d_in[6];
  const float* wih1f = (const float*)d_in[7];
  const float* whh1f = (const float*)d_in[8];
  const float* b1f   = (const float*)d_in[9];
  const float* wih1b = (const float*)d_in[10];
  const float* whh1b = (const float*)d_in[11];
  const float* b1b   = (const float*)d_in[12];
  const float* w1    = (const float*)d_in[13];
  const float* bias1 = (const float*)d_in[14];
  const float* w2    = (const float*)d_in[15];
  const float* bias2 = (const float*)d_in[16];
  float* out = (float*)d_out;

  // workspace carve-up (256B aligned slices, ~20 MB total)
  char* ws = (char*)d_ws;
  size_t off = 0;
  auto carve = [&](size_t bytes) -> void* {
    void* p = ws + off;
    off = (off + bytes + 255) & ~(size_t)255;
    return p;
  };
  bf16*  wih0f_bf = (bf16*)carve((size_t)G_DIM * I_DIM * sizeof(bf16));
  bf16*  wih0b_bf = (bf16*)carve((size_t)G_DIM * I_DIM * sizeof(bf16));
  bf16*  wih1f_bf = (bf16*)carve((size_t)G_DIM * H2 * sizeof(bf16));
  bf16*  wih1b_bf = (bf16*)carve((size_t)G_DIM * H2 * sizeof(bf16));
  bf16*  w1_bf    = (bf16*)carve((size_t)128 * H2 * sizeof(bf16));
  float* whhT0f   = (float*)carve((size_t)H_DIM * G_DIM * sizeof(float));
  float* whhT0b   = (float*)carve((size_t)H_DIM * G_DIM * sizeof(float));
  float* whhT1f   = (float*)carve((size_t)H_DIM * G_DIM * sizeof(float));
  float* whhT1b   = (float*)carve((size_t)H_DIM * G_DIM * sizeof(float));
  float* xg0f     = (float*)carve((size_t)A_DIM * G_DIM * sizeof(float));
  float* xg0b     = (float*)carve((size_t)A_DIM * G_DIM * sizeof(float));
  float* xg1f     = (float*)carve((size_t)A_DIM * G_DIM * sizeof(float));
  float* xg1b     = (float*)carve((size_t)A_DIM * G_DIM * sizeof(float));
  bf16*  xchunk_bf = (bf16*)carve((size_t)A_DIM * I_DIM * sizeof(bf16));
  bf16*  y_bf      = (bf16*)carve((size_t)A_DIM * H2 * sizeof(bf16));
  bf16*  z_bf      = (bf16*)carve((size_t)A_DIM * H2 * sizeof(bf16));
  float* hdn      = (float*)carve((size_t)A_DIM * 128 * sizeof(float));
  float* hc       = (float*)carve((size_t)4 * 512 * sizeof(float));

  // one-time (per call) weight prep
  k_cvt_bf16<<<256, 256, 0, stream>>>(wih0f, wih0f_bf, G_DIM * I_DIM);
  k_cvt_bf16<<<256, 256, 0, stream>>>(wih0b, wih0b_bf, G_DIM * I_DIM);
  k_cvt_bf16<<<256, 256, 0, stream>>>(wih1f, wih1f_bf, G_DIM * H2);
  k_cvt_bf16<<<256, 256, 0, stream>>>(wih1b, wih1b_bf, G_DIM * H2);
  k_cvt_bf16<<<64, 256, 0, stream>>>(w1, w1_bf, 128 * H2);
  k_transpose<<<256, 256, 0, stream>>>(whh0f, whhT0f, G_DIM, H_DIM);
  k_transpose<<<256, 256, 0, stream>>>(whh0b, whhT0b, G_DIM, H_DIM);
  k_transpose<<<256, 256, 0, stream>>>(whh1f, whhT1f, G_DIM, H_DIM);
  k_transpose<<<256, 256, 0, stream>>>(whh1b, whhT1b, G_DIM, H_DIM);
  k_zero<<<8, 256, 0, stream>>>(hc, 4 * 512);

  // chunk-pipelined scan over T
  const dim3 gblk(64);
  for (int t = 0; t < T_DIM; ++t) {
    const float* xt = x + (size_t)t * I_DIM;   // row a at xt + a*(T*I)
    // pack this chunk of x to bf16 (keeps WMMA loop conversion-free)
    k_cvt_rows<<<256, 256, 0, stream>>>(xt, T_DIM * I_DIM, xchunk_bf, A_DIM, I_DIM);
    // layer-0 input projections (both dirs), WMMA bf16
    k_gemm_wmma<<<dim3(G_DIM / 128, A_DIM / 16, 2), gblk, 0, stream>>>(
        xchunk_bf, xchunk_bf, I_DIM, wih0f_bf, wih0b_bf, I_DIM,
        b0f, b0b, xg0f, xg0b, G_DIM, I_DIM);
    // layer-0 recurrence (fwd + bwd workgroups)
    k_lstm_chunk<<<2, 1024, 0, stream>>>(xg0f, xg0b, whhT0f, whhT0b, hc, 0, y_bf);
    // layer-1 input projections from fresh y chunk
    k_gemm_wmma<<<dim3(G_DIM / 128, A_DIM / 16, 2), gblk, 0, stream>>>(
        y_bf, y_bf, H2, wih1f_bf, wih1b_bf, H2,
        b1f, b1b, xg1f, xg1b, G_DIM, H2);
    // layer-1 recurrence
    k_lstm_chunk<<<2, 1024, 0, stream>>>(xg1f, xg1b, whhT1f, whhT1b, hc, 2, z_bf);
  }

  // head: hdn = z @ w1^T + bias1 (WMMA), then logits + softmax
  k_gemm_wmma<<<dim3(128 / 128, A_DIM / 16, 1), gblk, 0, stream>>>(
      z_bf, z_bf, H2, w1_bf, w1_bf, H2, bias1, bias1, hdn, hdn, 128, H2);
  k_head<<<A_DIM, 32, 0, stream>>>(hdn, w2, bias2, out);
}
